// FusionLSTMClassifier_9835475108393
// MI455X (gfx1250) — matrix operand order repaired and checked by reading, not compiled, not run on hardware
//
#include <hip/hip_runtime.h>
#include <stdint.h>

// ---------------------------------------------------------------------------
// Problem constants (match reference)
// ---------------------------------------------------------------------------
#define B_   128
#define S_   512
#define F_   64
#define G_   256     // 4*F
#define H_   4
#define D_   64
#define HID_ 20
#define NC_  3

typedef __attribute__((ext_vector_type(2)))  float  v2f;
typedef __attribute__((ext_vector_type(8)))  float  v8f;
typedef __attribute__((ext_vector_type(16))) __bf16 v16bf;

// ---------------------------------------------------------------------------
// WMMA wrappers (CDNA5 / gfx1250)
// ---------------------------------------------------------------------------
__device__ __forceinline__ v8f wmma_f32x4(v2f a, v2f b, v8f c) {
  // D = A(16x4 f32) * B(4x16 f32) + C(16x16 f32)
  return __builtin_amdgcn_wmma_f32_16x16x4_f32(false, a, false, b, (short)0, c,
                                               false, false);
}
__device__ __forceinline__ v8f wmma_bf16x32(v16bf a, v16bf b, v8f c) {
  // D = A(16x32 bf16) * B(32x16 bf16) + C(16x16 f32)
  return __builtin_amdgcn_wmma_f32_16x16x32_bf16(false, a, false, b, (short)0,
                                                 c, false, false);
}

// ---------------------------------------------------------------------------
// CDNA5 async global->LDS DMA (ASYNCcnt-tracked), per-lane 16B transfer
// ---------------------------------------------------------------------------
__device__ __forceinline__ void async_g2l_b128(void* lds, const void* g) {
  const uint32_t lds_off = (uint32_t)(uintptr_t)lds;   // low 32b = LDS offset
  asm volatile("global_load_async_to_lds_b128 %0, %1, off"
               :: "v"(lds_off), "v"((uint64_t)(uintptr_t)g)
               : "memory");
}
__device__ __forceinline__ void wait_asynccnt0() {
  asm volatile("s_wait_asynccnt 0" ::: "memory");
}
__device__ __forceinline__ void wait_dscnt0() {
  asm volatile("s_wait_dscnt 0" ::: "memory");
}

// ---------------------------------------------------------------------------
// bf16 bit helpers (avoid dependence on __bf16 arithmetic support)
// ---------------------------------------------------------------------------
__device__ __forceinline__ unsigned short f2bf(float f) {
  union { float f; unsigned u; } v; v.f = f;
  unsigned u = v.u;
  u += 0x7fffu + ((u >> 16) & 1u);   // round-to-nearest-even
  return (unsigned short)(u >> 16);
}
__device__ __forceinline__ __bf16 bfbits(unsigned short u) {
  __bf16 r; __builtin_memcpy(&r, &u, 2); return r;
}

__device__ __forceinline__ float sigm_(float x) { return 1.0f / (1.0f + __expf(-x)); }
__device__ __forceinline__ float mish_(float v) {
  float sp = (v > 20.0f) ? v : log1pf(__expf(v));   // softplus
  return v * tanhf(sp);
}
__device__ __forceinline__ float rmax16(float v) {
  #pragma unroll
  for (int m = 1; m < 16; m <<= 1) v = fmaxf(v, __shfl_xor(v, m, 32));
  return v;
}
__device__ __forceinline__ float rsum16(float v) {
  #pragma unroll
  for (int m = 1; m < 16; m <<= 1) v += __shfl_xor(v, m, 32);
  return v;
}

// ---------------------------------------------------------------------------
// K1: SequenceInstanceNorm over time axis (per (b,f))
// ---------------------------------------------------------------------------
__global__ __launch_bounds__(F_) void norm_kernel(
    const float* __restrict__ x, const float* __restrict__ nw,
    const float* __restrict__ nb, float* __restrict__ xn) {
  const int b = blockIdx.x, f = threadIdx.x;
  const float* xp = x + (size_t)b * S_ * F_ + f;
  float s = 0.f, ss = 0.f;
  for (int t = 0; t < S_; ++t) { float v = xp[(size_t)t * F_]; s += v; ss += v * v; }
  const float mean = s * (1.0f / S_);
  const float var  = ss * (1.0f / S_) - mean * mean;
  const float inv  = rsqrtf(var + 1e-5f);
  const float w = nw[f], bb = nb[f];
  float* op = xn + (size_t)b * S_ * F_ + f;
  for (int t = 0; t < S_; ++t)
    op[(size_t)t * F_] = (xp[(size_t)t * F_] - mean) * inv * w + bb;
}

// ---------------------------------------------------------------------------
// K2/K4: generic GEMM  C[M,N] = A[M,64] * W[N,64]^T + bias1 + bias2
// fp32 WMMA 16x16x4.  MODE 0: f32 [M,N] | 1: bf16 [M,N] | 2: bf16 vT[b,h,d,s]
// block = 256 thr (8 waves), wave -> one 16-row M tile; grid.y -> 64-col group
// ---------------------------------------------------------------------------
template <int MODE>
__global__ __launch_bounds__(256) void gemm_k64(
    const float* __restrict__ A, const float* __restrict__ W,
    const float* __restrict__ bias1, const float* __restrict__ bias2,
    void* __restrict__ out, int M, int N) {
  const int wave = threadIdx.x >> 5;
  const int lane = threadIdx.x & 31;
  const int m16  = lane & 15;
  const int half = lane >> 4;
  const int mtile = blockIdx.x * 8 + wave;
  const int nbase = blockIdx.y * 64;

  const float* Arow = A + (size_t)(mtile * 16 + m16) * 64;
  v8f acc[4];
  #pragma unroll
  for (int nt = 0; nt < 4; ++nt) acc[nt] = (v8f)0.0f;

  #pragma unroll
  for (int ks = 0; ks < 16; ++ks) {
    const int k0 = ks * 4 + half * 2;            // K = j + 2*half + 4*ks
    v2f a; a.x = Arow[k0]; a.y = Arow[k0 + 1];
    #pragma unroll
    for (int nt = 0; nt < 4; ++nt) {
      const float* Wrow = W + (size_t)(nbase + nt * 16 + m16) * 64;
      v2f b; b.x = Wrow[k0]; b.y = Wrow[k0 + 1];
      acc[nt] = wmma_f32x4(a, b, acc[nt]);
    }
  }

  #pragma unroll
  for (int nt = 0; nt < 4; ++nt) {
    #pragma unroll
    for (int j = 0; j < 8; ++j) {
      const int row = mtile * 16 + j + 8 * half;
      const int col = nbase + nt * 16 + m16;
      float v = acc[nt][j];
      if (bias1) v += bias1[col];
      if (bias2) v += bias2[col];
      if (MODE == 0) {
        ((float*)out)[(size_t)row * N + col] = v;
      } else if (MODE == 1) {
        ((unsigned short*)out)[(size_t)row * N + col] = f2bf(v);
      } else {
        const int b = row >> 9, s = row & (S_ - 1);   // S_ = 512
        const int h = col >> 6, d = col & 63;
        ((unsigned short*)out)[((size_t)(b * H_ + h) * D_ + d) * S_ + s] = f2bf(v);
      }
    }
  }
}

// ---------------------------------------------------------------------------
// K3: Mish-LSTM sequential scan. 8 blocks x 16 batch rows, 512 thr (16 waves).
// Each step: gates(16x256) = gx tile + h(16x64) * wh^T  via fp32 WMMA.
// gx tiles are double-buffered in LDS via CDNA5 async global->LDS DMA so the
// 16KB/step fetch overlaps the recurrence (the latency-critical section).
// ---------------------------------------------------------------------------
__global__ __launch_bounds__(512) void lstm_scan(
    const float* __restrict__ gx, const float* __restrict__ wh,
    float* __restrict__ enc) {
  __shared__ __align__(16) float gxbuf[2][16][G_];   // 32 KB, async DMA dest
  __shared__ float gates[16][G_ + 8];                // 16.9 KB
  __shared__ float h_lds[16][F_];
  __shared__ float c_lds[16][F_];

  const int tid  = threadIdx.x;
  const int wave = tid >> 5;
  const int lane = tid & 31;
  const int m16  = lane & 15;
  const int half = lane >> 4;
  const int rowbase = blockIdx.x * 16;

  for (int idx = tid; idx < 16 * F_; idx += 512) {
    h_lds[idx >> 6][idx & 63] = 0.f;
    c_lds[idx >> 6][idx & 63] = 0.f;
  }

  // prime the pipeline: async-load gx tile for t = 0
  // tile = 16 rows x 256 f32 = 1024 x 16B chunks; 2 chunks per thread
  #pragma unroll
  for (int r = 0; r < 2; ++r) {
    const int idx = tid + r * 512;
    const int m = idx >> 6, c4 = (idx & 63) * 4;
    async_g2l_b128(&gxbuf[0][m][c4],
                   gx + ((size_t)(rowbase + m) * S_ + 0) * G_ + c4);
  }
  wait_asynccnt0();
  __syncthreads();

  for (int t = 0; t < S_; ++t) {
    const int cur = t & 1, nxt = cur ^ 1;

    // prefetch next timestep's gx tile (overlaps with WMMA below)
    if (t + 1 < S_) {
      #pragma unroll
      for (int r = 0; r < 2; ++r) {
        const int idx = tid + r * 512;
        const int m = idx >> 6, c4 = (idx & 63) * 4;
        async_g2l_b128(&gxbuf[nxt][m][c4],
                       gx + ((size_t)(rowbase + m) * S_ + (t + 1)) * G_ + c4);
      }
    }

    // gates = gxbuf[cur] + h * wh^T   (wave 'w' owns N-tile w)
    {
      const int nt = wave;
      v8f acc = (v8f)0.0f;
      #pragma unroll
      for (int ks = 0; ks < 16; ++ks) {
        const int k0 = ks * 4 + half * 2;
        v2f a; a.x = h_lds[m16][k0]; a.y = h_lds[m16][k0 + 1];
        const float* Wrow = wh + (size_t)(nt * 16 + m16) * 64;
        v2f b; b.x = Wrow[k0]; b.y = Wrow[k0 + 1];
        acc = wmma_f32x4(a, b, acc);
      }
      #pragma unroll
      for (int j = 0; j < 8; ++j) {
        const int r = j + 8 * half, c = nt * 16 + m16;
        gates[r][c] = gxbuf[cur][r][c] + acc[j];
      }
    }
    __syncthreads();

    // elementwise cell update
    for (int idx = tid; idx < 16 * F_; idx += 512) {
      const int m = idx >> 6, f = idx & 63;
      const float iv = sigm_(gates[m][f]);
      const float fv = sigm_(gates[m][64 + f]);
      const float gv = mish_(gates[m][128 + f]);
      const float ov = sigm_(gates[m][192 + f]);
      const float c  = fv * c_lds[m][f] + iv * gv;
      const float h  = ov * mish_(c);
      c_lds[m][f] = c;
      h_lds[m][f] = h;
      enc[((size_t)(rowbase + m) * S_ + t) * F_ + f] = h;
    }
    wait_asynccnt0();       // next tile resident before anyone reads it
    __syncthreads();
  }
}

// ---------------------------------------------------------------------------
// K7: flash attention per (b,h); bf16 WMMA; accumulates column-sums of att
// directly into attsum[b, h*64+d] (pooled mean is linear).
// block = 256 thr (8 waves); wave -> 16-query tile; grid (qblk=4, h=4, b=128)
// The P-tile C-layout -> A-layout conversion round-trips through wave-private
// LDS; DS ops are in-order per wave so s_wait_dscnt 0 suffices (no barriers).
// ---------------------------------------------------------------------------
__global__ __launch_bounds__(256) void attn_pool(
    const unsigned short* __restrict__ q, const unsigned short* __restrict__ k,
    const unsigned short* __restrict__ vT, float* __restrict__ attsum) {
  __shared__ unsigned short p_lds[8][16][32];  // per-wave P tile (bf16 bits)

  const int tid  = threadIdx.x;
  const int wave = tid >> 5;
  const int lane = tid & 31;
  const int m16  = lane & 15;
  const int half = lane >> 4;
  const int b = blockIdx.z, h = blockIdx.y, qblk = blockIdx.x;
  const int qbase = (qblk * 8 + wave) * 16;

  const unsigned short* Q = q + ((size_t)b * S_) * (H_ * D_) + h * D_;
  const unsigned short* K = k + ((size_t)b * S_) * (H_ * D_) + h * D_;
  const unsigned short* V = vT + (size_t)(b * H_ + h) * D_ * S_;

  // preload two 16x32 Q fragments (d 0..31, 32..63)
  v16bf qa[2];
  const unsigned short* qrow = Q + (size_t)(qbase + m16) * (H_ * D_);
  #pragma unroll
  for (int kc = 0; kc < 2; ++kc) {
    #pragma unroll
    for (int j = 0; j < 8; ++j) {
      const int base = (j < 4) ? 2 * j : 8 + 2 * j;   // 16-bit A interleave
      const int d = kc * 32 + base + 8 * half;
      qa[kc][2 * j]     = bfbits(qrow[d]);
      qa[kc][2 * j + 1] = bfbits(qrow[d + 1]);
    }
  }

  float mst[8], lst[8];
  v8f o[4];
  #pragma unroll
  for (int j = 0; j < 8; ++j) { mst[j] = -3.0e38f; lst[j] = 0.f; }
  #pragma unroll
  for (int nt = 0; nt < 4; ++nt) o[nt] = (v8f)0.0f;

  for (int kb = 0; kb < S_; kb += 32) {
    // ---- scores: two 16-key groups, K-dim = 64 (2 bf16 WMMAs each) ----
    v8f sc[2];
    #pragma unroll
    for (int g = 0; g < 2; ++g) {
      sc[g] = (v8f)0.0f;
      const unsigned short* krow = K + (size_t)(kb + g * 16 + m16) * (H_ * D_);
      #pragma unroll
      for (int kc = 0; kc < 2; ++kc) {
        v16bf bb;
        const int dd = kc * 32 + 8 * half;
        #pragma unroll
        for (int i = 0; i < 16; ++i) bb[i] = bfbits(krow[dd + i + (i >= 8 ? 8 : 0)]);
        sc[g] = wmma_bf16x32(qa[kc], bb, sc[g]);
      }
      sc[g] = sc[g] * 0.125f;   // D^-0.5
    }

    // ---- online softmax over this 32-key chunk ----
    float al[8];
    #pragma unroll
    for (int j = 0; j < 8; ++j) {
      float mx = rmax16(fmaxf(sc[0][j], sc[1][j]));
      const float nm = fmaxf(mst[j], mx);
      al[j] = __expf(mst[j] - nm);
      const float p0 = __expf(sc[0][j] - nm);
      const float p1 = __expf(sc[1][j] - nm);
      sc[0][j] = p0; sc[1][j] = p1;
      lst[j] = lst[j] * al[j] + rsum16(p0 + p1);
      mst[j] = nm;
    }
    #pragma unroll
    for (int nt = 0; nt < 4; ++nt)
      #pragma unroll
      for (int j = 0; j < 8; ++j) o[nt][j] *= al[j];

    // ---- P tile: C-layout regs -> wave-private LDS -> 16x32 bf16 A frag ----
    #pragma unroll
    for (int j = 0; j < 8; ++j) {
      const int r = j + 8 * half;
      p_lds[wave][r][m16]      = f2bf(sc[0][j]);
      p_lds[wave][r][16 + m16] = f2bf(sc[1][j]);
    }
    wait_dscnt0();
    v16bf pa;
    #pragma unroll
    for (int j = 0; j < 8; ++j) {
      const int base = (j < 4) ? 2 * j : 8 + 2 * j;
      const int kk = base + 8 * half;
      pa[2 * j]     = bfbits(p_lds[wave][m16][kk]);
      pa[2 * j + 1] = bfbits(p_lds[wave][m16][kk + 1]);
    }
    wait_dscnt0();

    // ---- O += P(16x32) * V(32x64): 4 d-tiles ----
    #pragma unroll
    for (int nt = 0; nt < 4; ++nt) {
      const unsigned short* vrow = V + (size_t)(nt * 16 + m16) * S_ + kb + 8 * half;
      v16bf bv;
      #pragma unroll
      for (int i = 0; i < 16; ++i) bv[i] = bfbits(vrow[i + (i >= 8 ? 8 : 0)]);
      o[nt] = wmma_bf16x32(pa, bv, o[nt]);
    }
  }

  // ---- normalize by l, reduce over the 16 query rows, accumulate ----
  float inv[8];
  #pragma unroll
  for (int j = 0; j < 8; ++j) inv[j] = 1.0f / lst[j];
  #pragma unroll
  for (int nt = 0; nt < 4; ++nt) {
    float cs = 0.f;
    #pragma unroll
    for (int j = 0; j < 8; ++j) cs += o[nt][j] * inv[j];
    cs += __shfl_xor(cs, 16, 32);        // rows 0-7 + rows 8-15
    if (half == 0)
      atomicAdd(&attsum[(size_t)b * (H_ * D_) + h * D_ + nt * 16 + m16], cs);
  }
}

// ---------------------------------------------------------------------------
// K8: pooled[b,f] = mean_s enc + (mean_s att) . wo[f,:] + bo[f]
// ---------------------------------------------------------------------------
__global__ __launch_bounds__(F_) void pool_kernel(
    const float* __restrict__ enc, const float* __restrict__ attsum,
    const float* __restrict__ wo, const float* __restrict__ bo,
    float* __restrict__ pooled) {
  const int b = blockIdx.x, f = threadIdx.x;
  const float* e = enc + (size_t)b * S_ * F_ + f;
  float s = 0.f;
  for (int t = 0; t < S_; ++t) s += e[(size_t)t * F_];
  const float* as = attsum + (size_t)b * (H_ * D_);
  const float* w  = wo + (size_t)f * (H_ * D_);
  float dot = 0.f;
  for (int e2 = 0; e2 < H_ * D_; ++e2) dot += as[e2] * w[e2];
  pooled[b * F_ + f] = (s + dot) * (1.0f / S_) + bo[f];
}

// ---------------------------------------------------------------------------
// K9: classifier MLP  out = mish(pooled*w1^T+b1)*w2^T+b2
// ---------------------------------------------------------------------------
__global__ __launch_bounds__(128) void cls_kernel(
    const float* __restrict__ pooled, const float* __restrict__ w1,
    const float* __restrict__ b1, const float* __restrict__ w2,
    const float* __restrict__ b2, float* __restrict__ out) {
  const int b = threadIdx.x;
  float p[F_];
  #pragma unroll
  for (int i = 0; i < F_; ++i) p[i] = pooled[b * F_ + i];
  float hid[HID_];
  for (int j = 0; j < HID_; ++j) {
    float a = b1[j];
    #pragma unroll
    for (int i = 0; i < F_; ++i) a += p[i] * w1[j * F_ + i];
    hid[j] = mish_(a);
  }
  for (int c = 0; c < NC_; ++c) {
    float a = b2[c];
    #pragma unroll
    for (int j = 0; j < HID_; ++j) a += hid[j] * w2[c * HID_ + j];
    out[b * NC_ + c] = a;
  }
}

// ---------------------------------------------------------------------------
extern "C" void kernel_launch(void* const* d_in, const int* in_sizes, int n_in,
                              void* d_out, int out_size, void* d_ws, size_t ws_size,
                              hipStream_t stream) {
  (void)in_sizes; (void)n_in; (void)out_size; (void)ws_size;
  const float* x      = (const float*)d_in[0];
  const float* norm_w = (const float*)d_in[1];
  const float* norm_b = (const float*)d_in[2];
  const float* wi     = (const float*)d_in[3];
  const float* bi     = (const float*)d_in[4];
  const float* wh     = (const float*)d_in[5];
  const float* bh     = (const float*)d_in[6];
  const float* wq     = (const float*)d_in[7];
  const float* bq     = (const float*)d_in[8];
  const float* wk     = (const float*)d_in[9];
  const float* bk     = (const float*)d_in[10];
  const float* wv     = (const float*)d_in[11];
  const float* bv     = (const float*)d_in[12];
  const float* wo     = (const float*)d_in[13];
  const float* bo     = (const float*)d_in[14];
  const float* w1     = (const float*)d_in[15];
  const float* b1     = (const float*)d_in[16];
  const float* w2     = (const float*)d_in[17];
  const float* b2     = (const float*)d_in[18];

  // workspace carving
  char* ws = (char*)d_ws;
  size_t off = 0;
  const size_t MS = (size_t)B_ * S_;               // 65536 rows
  float*          xn     = (float*)(ws + off); off += MS * F_ * 4;     // 16.8 MB
  float*          gx     = (float*)(ws + off); off += MS * G_ * 4;     // 67.1 MB
  float*          enc    = (float*)(ws + off); off += MS * F_ * 4;     // 16.8 MB
  unsigned short* qb     = (unsigned short*)(ws + off); off += MS * (H_*D_) * 2;
  unsigned short* kb     = (unsigned short*)(ws + off); off += MS * (H_*D_) * 2;
  unsigned short* vT     = (unsigned short*)(ws + off); off += MS * (H_*D_) * 2;
  float*          attsum = (float*)(ws + off); off += (size_t)B_ * (H_*D_) * 4;
  float*          pooled = (float*)(ws + off); off += (size_t)B_ * F_ * 4;

  hipMemsetAsync(attsum, 0, (size_t)B_ * (H_ * D_) * sizeof(float), stream);

  // 1) instance norm
  norm_kernel<<<B_, F_, 0, stream>>>(x, norm_w, norm_b, xn);
  // 2) gx = xn*wi^T + bi + bh
  gemm_k64<0><<<dim3(MS / 128, G_ / 64), 256, 0, stream>>>(xn, wi, bi, bh, gx,
                                                           (int)MS, G_);
  // 3) LSTM scan (async double-buffered gx staging)
  lstm_scan<<<B_ / 16, 512, 0, stream>>>(gx, wh, enc);
  // 4) Q/K/V projections (bf16), V transposed per head
  gemm_k64<1><<<dim3(MS / 128, 4), 256, 0, stream>>>(enc, wq, bq, nullptr, qb,
                                                     (int)MS, H_ * D_);
  gemm_k64<1><<<dim3(MS / 128, 4), 256, 0, stream>>>(enc, wk, bk, nullptr, kb,
                                                     (int)MS, H_ * D_);
  gemm_k64<2><<<dim3(MS / 128, 4), 256, 0, stream>>>(enc, wv, bv, nullptr, vT,
                                                     (int)MS, H_ * D_);
  // 5) flash attention + pooled accumulation
  attn_pool<<<dim3(4, H_, B_), 256, 0, stream>>>(qb, kb, vT, attsum);
  // 6) pool + output projection (folded), 7) classifier
  pool_kernel<<<B_, F_, 0, stream>>>(enc, attsum, wo, bo, pooled);
  cls_kernel<<<1, 128, 0, stream>>>(pooled, w1, b1, w2, b2, (float*)d_out);
}
